// TorchParametricSolver_81767587381597
// MI455X (gfx1250) — compile-verified, matrix operand order checked by reading, not run on hardware
//
#include <hip/hip_runtime.h>
#include <stdint.h>

typedef __attribute__((ext_vector_type(16))) _Float16 v16h;
typedef __attribute__((ext_vector_type(8)))  float    v8f;

#define N1 2359296      // total labels / memory slots
#define NR 294912       // N_ROWS
#define N2 262144       // N_OPS
#define T1 576          // sort-1 tiles (N1 / 4096)
#define T2 64           // sort-2 tiles (N2 / 4096)
#define RS_E 16
#define RS_TILE 4096

// ---------------- radix sort (stable LSD, 4 x 8-bit) ----------------

__global__ void k_sort_init(const float* __restrict__ src, unsigned* __restrict__ keys,
                            unsigned* __restrict__ vals, int n) {
  int i = blockIdx.x * blockDim.x + threadIdx.x;
  if (i >= n) return;
  unsigned u = __float_as_uint(src[i]);
  u ^= (u & 0x80000000u) ? 0xFFFFFFFFu : 0x80000000u;   // orderable uint
  keys[i] = u;
  vals[i] = (unsigned)i;
}

__global__ void k_sort_hist(const unsigned* __restrict__ keys, unsigned* __restrict__ hist,
                            int shift, int ntiles) {
  __shared__ unsigned cnt[256];
  int t = threadIdx.x;
  cnt[t] = 0;
  __syncthreads();
  const unsigned* kp = keys + (size_t)blockIdx.x * RS_TILE;
#pragma unroll
  for (int e = 0; e < RS_E; e++) {
    unsigned d = (kp[t * RS_E + e] >> shift) & 255u;
    atomicAdd(&cnt[d], 1u);
  }
  __syncthreads();
  hist[(size_t)t * ntiles + blockIdx.x] = cnt[t];       // digit-major layout
}

__global__ void k_scan(unsigned* __restrict__ data, int total) {
  __shared__ unsigned s[1024];
  int t = threadIdx.x;
  int seg = (total + 1023) >> 10;
  int base = t * seg;
  unsigned sum = 0;
  for (int i = 0; i < seg; i++) { int idx = base + i; if (idx < total) sum += data[idx]; }
  s[t] = sum;
  __syncthreads();
  for (int off = 1; off < 1024; off <<= 1) {
    unsigned v = (t >= off) ? s[t - off] : 0u;
    __syncthreads();
    s[t] += v;
    __syncthreads();
  }
  unsigned run = s[t] - sum;                            // exclusive start of my segment
  for (int i = 0; i < seg; i++) {
    int idx = base + i;
    if (idx < total) { unsigned v = data[idx]; data[idx] = run; run += v; }
  }
}

__global__ void k_sort_scatter(const unsigned* __restrict__ keysIn, const unsigned* __restrict__ valsIn,
                               unsigned* __restrict__ keysOut, unsigned* __restrict__ valsOut,
                               const unsigned* __restrict__ histS, int shift, int ntiles) {
  __shared__ unsigned lbuf[RS_TILE];
  __shared__ unsigned ssum[256];
  __shared__ unsigned cnt[256];
  __shared__ unsigned dstart[256];
  int t = threadIdx.x;
  size_t tb = (size_t)blockIdx.x * RS_TILE;
  unsigned key[RS_E], val[RS_E], pos[RS_E];
#pragma unroll
  for (int e = 0; e < RS_E; e++) {
    key[e] = keysIn[tb + t * RS_E + e];
    val[e] = valsIn[tb + t * RS_E + e];
  }
  // 8 stable binary splits on bits [shift, shift+8)
  for (int s = 0; s < 8; s++) {
    int bp = shift + s;
    unsigned bits = 0, b[RS_E];
#pragma unroll
    for (int e = 0; e < RS_E; e++) { b[e] = (key[e] >> bp) & 1u; bits += b[e]; }
    ssum[t] = bits;
    __syncthreads();
    for (int off = 1; off < 256; off <<= 1) {
      unsigned v = (t >= off) ? ssum[t - off] : 0u;
      __syncthreads();
      ssum[t] += v;
      __syncthreads();
    }
    unsigned excl = ssum[t] - bits;
    unsigned totalOnes = ssum[255];
    unsigned totalFalses = RS_TILE - totalOnes;
    unsigned run = excl;
#pragma unroll
    for (int e = 0; e < RS_E; e++) {
      unsigned i = t * RS_E + e;
      pos[e] = b[e] ? (totalFalses + run) : (i - run);
      run += b[e];
    }
    __syncthreads();
#pragma unroll
    for (int e = 0; e < RS_E; e++) lbuf[pos[e]] = key[e];
    __syncthreads();
#pragma unroll
    for (int e = 0; e < RS_E; e++) key[e] = lbuf[t * RS_E + e];
    __syncthreads();
#pragma unroll
    for (int e = 0; e < RS_E; e++) lbuf[pos[e]] = val[e];
    __syncthreads();
#pragma unroll
    for (int e = 0; e < RS_E; e++) val[e] = lbuf[t * RS_E + e];
    __syncthreads();
  }
  // local digit offsets -> global scatter
  cnt[t] = 0;
  __syncthreads();
#pragma unroll
  for (int e = 0; e < RS_E; e++) atomicAdd(&cnt[(key[e] >> shift) & 255u], 1u);
  __syncthreads();
  unsigned c = cnt[t];
  ssum[t] = c;
  __syncthreads();
  for (int off = 1; off < 256; off <<= 1) {
    unsigned v = (t >= off) ? ssum[t - off] : 0u;
    __syncthreads();
    ssum[t] += v;
    __syncthreads();
  }
  dstart[t] = ssum[t] - c;
  __syncthreads();
#pragma unroll
  for (int e = 0; e < RS_E; e++) {
    unsigned i = t * RS_E + e;
    unsigned d = (key[e] >> shift) & 255u;
    unsigned g = histS[(size_t)d * ntiles + blockIdx.x] + (i - dstart[d]);
    keysOut[g] = key[e];
    valsOut[g] = val[e];
  }
}

// ---------------- conv(1->8) + ReLU + permutation scatter ----------------

__global__ void k_conv_scatter(const unsigned* __restrict__ addr, int off, int h, int w,
                               const float* __restrict__ cw, const float* __restrict__ cb,
                               float* __restrict__ memb) {
  __shared__ float wsm[72];
  __shared__ float bsm[8];
  int t = threadIdx.x;
  if (t < 72) wsm[t] = cw[t];
  if (t < 8)  bsm[t] = cb[t];
  __syncthreads();
  int p = blockIdx.x * 256 + t;
  int r = p / w, c = p % w;
  float acc[8];
#pragma unroll
  for (int oc = 0; oc < 8; oc++) acc[oc] = bsm[oc];
#pragma unroll
  for (int ky = 0; ky < 3; ky++) {
    int rr = r + ky - 1;
#pragma unroll
    for (int kx = 0; kx < 3; kx++) {
      int cc = c + kx - 1;
      float x = (rr >= 0 && rr < h && cc >= 0 && cc < w) ? (float)addr[off + rr * w + cc] : 0.f;
#pragma unroll
      for (int oc = 0; oc < 8; oc++) acc[oc] += x * wsm[oc * 9 + ky * 3 + kx];
    }
  }
  unsigned a = addr[off + p];
#pragma unroll
  for (int oc = 0; oc < 8; oc++) {
    float v = acc[oc];
    memb[(size_t)oc * N1 + a] = (v > 0.f) ? v : 0.f;
  }
}

// ---------------- conv(8->16) + ReLU + fused adaptive-avg-pool partials ----------------

__global__ void k_conv2_pool(const float* __restrict__ memb, const float* __restrict__ mw,
                             const float* __restrict__ mb, float* __restrict__ partial) {
  __shared__ float wsm[1152];
  __shared__ float bsm[16];
  __shared__ float red[256 * 16];
  int t = threadIdx.x;
  for (int i = t; i < 1152; i += 256) wsm[i] = mw[i];
  if (t < 16) bsm[t] = mb[t];
  __syncthreads();
  int r0 = blockIdx.x * 128;         // 128 rows/block; 128 | 73728 so rb is constant per block
  int c = t & 7;
  int rb0 = t >> 3;                  // 0..31
  float pacc[16];
#pragma unroll
  for (int o = 0; o < 16; o++) pacc[o] = 0.f;
  for (int s = 0; s < 4; s++) {
    int r = r0 + rb0 + 32 * s;
    float acc[16];
#pragma unroll
    for (int o = 0; o < 16; o++) acc[o] = bsm[o];
#pragma unroll
    for (int ic = 0; ic < 8; ic++) {
#pragma unroll
      for (int ky = 0; ky < 3; ky++) {
        int rr = r + ky - 1;
        bool rv = (rr >= 0 && rr < NR);
#pragma unroll
        for (int kx = 0; kx < 3; kx++) {
          int cc = c + kx - 1;
          float x = (rv && cc >= 0 && cc < 8) ? memb[(size_t)ic * N1 + (size_t)rr * 8 + cc] : 0.f;
#pragma unroll
          for (int o = 0; o < 16; o++) acc[o] += x * wsm[((o * 8 + ic) * 3 + ky) * 3 + kx];
        }
      }
    }
#pragma unroll
    for (int o = 0; o < 16; o++) { float v = acc[o]; pacc[o] += (v > 0.f) ? v : 0.f; }
  }
#pragma unroll
  for (int o = 0; o < 16; o++) red[t * 16 + o] = pacc[o];
  __syncthreads();
  if (t < 64) {                       // deterministic fixed-order reduction
    int cb = t >> 4, ch = t & 15;     // cb = column-pair block (c>>1), 4 of them
    float sacc = 0.f;
    for (int k = 0; k < 32; k++) {
      sacc += red[((2 * cb)     + 8 * k) * 16 + ch];
      sacc += red[((2 * cb + 1) + 8 * k) * 16 + ch];
    }
    partial[(size_t)blockIdx.x * 64 + t] = sacc;
  }
}

__global__ void k_pool_fin(const float* __restrict__ partial, float* __restrict__ pooled) {
  int t = threadIdx.x;                // 256 threads -> flat index ch*16 + rb*4 + cb
  int ch = t >> 4, rb = (t >> 2) & 3, cb = t & 3;
  float s = 0.f;
  for (int q = 0; q < 576; q++)
    s += partial[(size_t)(rb * 576 + q) * 64 + cb * 16 + ch];
  pooled[t] = s * (1.0f / 147456.0f); // 73728 rows * 2 cols per pooled cell
}

// ---------------- projection GEMV via v_wmma_f32_16x16x32_f16 ----------------
// D[m][n] = sum_k A[m][k]*B[k][n]; all 16 A-rows broadcast the pooled chunk,
// B columns hold 16 different proj_w rows -> row 0 of D gives 16 logits.

__global__ void k_proj_wmma(const float* __restrict__ pooled, const float* __restrict__ pw,
                            const float* __restrict__ pb, float* __restrict__ out) {
  __shared__ float pl[256];
  __shared__ float wl[4][512];
  int t = threadIdx.x;                // 128 threads = 4 waves
  for (int i = t; i < 256; i += 128) pl[i] = pooled[i];
  int wave = t >> 5, lane = t & 31;
  int tile = blockIdx.x * 4 + wave;
  int base = tile * 16;               // 16 outputs per wave
  int g = lane >> 4, n = lane & 15;
  v8f acc = {};
  __syncthreads();
  for (int cch = 0; cch < 8; cch++) { // 256 = 8 chunks of K=32
    for (int idx = lane; idx < 512; idx += 32) {       // stage 16 rows x 32 cols (f32)
      int rrow = idx >> 5, k = idx & 31;
      wl[wave][idx] = pw[(size_t)(base + rrow) * 256 + cch * 32 + k];
    }
    if (cch < 7)
      __builtin_prefetch(&pw[(size_t)base * 256 + (cch + 1) * 32], 0, 1); // global_prefetch_b8
    __syncthreads();
    v16h af, bf;
#pragma unroll
    for (int hh = 0; hh < 16; hh++) { // ISA 16-bit A layout: VGPR j holds K pair; lane group g offsets K by 8/ +16 for j>=4
      int j = hh >> 1, p = hh & 1;
      int kk = (j < 4) ? ((g ? 8 : 0) + 2 * j + p)
                       : (16 + (g ? 8 : 0) + 2 * (j - 4) + p);
      af[hh] = (_Float16)pl[cch * 32 + kk];            // broadcast pooled chunk to all M rows
      bf[hh] = (_Float16)wl[wave][n * 32 + kk];        // column n = weight row base+n
    }
    acc = __builtin_amdgcn_wmma_f32_16x16x32_f16(false, af, false, bf,
                                                 (short)0, acc, false, false);
    __syncthreads();
  }
  if (lane < 16)                                       // D row 0: VGPR0, lanes 0..15
    out[base + lane] = acc[0] + pb[base + lane];
}

// ---------------- staged hop penalties ----------------

__device__ __forceinline__ float stagedTerm(float h, int e) {
  float base = (e == 1) ? h : h * h;
  float m = (h <= 2.f) ? 1.f : (h <= 4.f) ? 1.5f : (h <= 8.f) ? 2.f : (h <= 16.f) ? 3.f : 5.f;
  return base * m;
}
__device__ __forceinline__ float penPair(float d) {
  float hp = (d > 0.f) ? d : 0.f;
  float hn = (-d > 0.f) ? -d : 0.f;
  return stagedTerm(hp, 1) + stagedTerm(hn, 2);
}

__global__ void k_penalty(const unsigned* __restrict__ order, const unsigned* __restrict__ addr,
                          float* __restrict__ psum) {
  __shared__ float si[256], sa[256];
  int t = threadIdx.x;
  int p = blockIdx.x * 256 + t;
  unsigned kp = order[p];
  unsigned i = kp >> 9, j = kp & 511u;                 // TH=TW=512
  float a0 = (float)addr[i * 1024u + j];
  float a1 = (float)addr[1048576u + i * 1024u + j];
  float a2 = (float)addr[2097152u + kp];
  float intra = penPair(a1 - a0) + penPair(a2 - a1);
  float inter = 0.f;
  if (p > 0) {
    unsigned kq = order[p - 1];
    float b2 = (float)addr[2097152u + kq];             // previous op's dst
    inter = penPair(a0 - b2);
  }
  si[t] = inter; sa[t] = intra;
  __syncthreads();
  for (int s = 128; s > 0; s >>= 1) {
    if (t < s) { si[t] += si[t + s]; sa[t] += sa[t + s]; }
    __syncthreads();
  }
  if (t == 0) { psum[blockIdx.x * 2] = si[0]; psum[blockIdx.x * 2 + 1] = sa[0]; }
}

__global__ void k_pen_fin(const float* __restrict__ psum, float* __restrict__ out2) {
  __shared__ float si[256], sa[256];
  int t = threadIdx.x;
  float a = 0.f, b = 0.f;
  for (int idx = t; idx < 1024; idx += 256) { a += psum[idx * 2]; b += psum[idx * 2 + 1]; }
  si[t] = a; sa[t] = b;
  __syncthreads();
  for (int s = 128; s > 0; s >>= 1) {
    if (t < s) { si[t] += si[t + s]; sa[t] += sa[t + s]; }
    __syncthreads();
  }
  if (t == 0) { out2[0] = si[0]; out2[1] = sa[0]; }
}

// ---------------- launch ----------------

extern "C" void kernel_launch(void* const* d_in, const int* in_sizes, int n_in,
                              void* d_out, int out_size, void* d_ws, size_t ws_size,
                              hipStream_t stream) {
  const float* mem_logits = (const float*)d_in[0];
  const float* cw0 = (const float*)d_in[1]; const float* cb0 = (const float*)d_in[2];
  const float* cw1 = (const float*)d_in[3]; const float* cb1 = (const float*)d_in[4];
  const float* cw2 = (const float*)d_in[5]; const float* cb2 = (const float*)d_in[6];
  const float* mw  = (const float*)d_in[7]; const float* mb  = (const float*)d_in[8];
  const float* pw  = (const float*)d_in[9]; const float* pb  = (const float*)d_in[10];
  float* out = (float*)d_out;

  char* ws = (char*)d_ws;
  size_t cur = 0;
  auto alloc = [&](size_t bytes) { size_t o = cur; cur = (cur + bytes + 255) & ~(size_t)255; return o; };
  unsigned* keyA = (unsigned*)(ws + alloc((size_t)N1 * 4));
  unsigned* keyB = (unsigned*)(ws + alloc((size_t)N1 * 4));
  unsigned* valA = (unsigned*)(ws + alloc((size_t)N1 * 4));
  unsigned* valB = (unsigned*)(ws + alloc((size_t)N1 * 4));
  unsigned* hist = (unsigned*)(ws + alloc((size_t)256 * T1 * 4));
  float* memb    = (float*)   (ws + alloc((size_t)8 * N1 * 4));
  float* partial = (float*)   (ws + alloc((size_t)2304 * 64 * 4));
  float* pooled  = (float*)   (ws + alloc(256 * 4));
  float* psum    = (float*)   (ws + alloc((size_t)1024 * 2 * 4));
  // second sort reuses the (then-dead) memb region
  unsigned* k2A = (unsigned*)memb;
  unsigned* k2B = k2A + N2;
  unsigned* v2A = k2B + N2;
  unsigned* v2B = v2A + N2;

  // ---- sort 1: addr = argsort(mem_logits) ----
  k_sort_init<<<N1 / 256, 256, 0, stream>>>(mem_logits, keyA, valA, N1);
  unsigned *ki = keyA, *vi = valA, *ko = keyB, *vo = valB;
  for (int pass = 0; pass < 4; pass++) {
    int shift = pass * 8;
    k_sort_hist<<<T1, 256, 0, stream>>>(ki, hist, shift, T1);
    k_scan<<<1, 1024, 0, stream>>>(hist, 256 * T1);
    k_sort_scatter<<<T1, 256, 0, stream>>>(ki, vi, ko, vo, hist, shift, T1);
    unsigned* tk = ki; ki = ko; ko = tk;
    unsigned* tv = vi; vi = vo; vo = tv;
  }
  unsigned* addr = vi;  // ends in valA

  // ---- conv(1->8)+scatter for the three matrices ----
  k_conv_scatter<<<4096, 256, 0, stream>>>(addr, 0,       1024, 1024, cw0, cb0, memb);
  k_conv_scatter<<<4096, 256, 0, stream>>>(addr, 1048576, 1024, 1024, cw1, cb1, memb);
  k_conv_scatter<<<1024, 256, 0, stream>>>(addr, 2097152, 512,  512,  cw2, cb2, memb);

  // ---- conv(8->16)+pool, finalize pooled[256] ----
  k_conv2_pool<<<2304, 256, 0, stream>>>(memb, mw, mb, partial);
  k_pool_fin<<<1, 256, 0, stream>>>(partial, pooled);

  // ---- WMMA projection: op_logits -> d_out[0..N2) ----
  k_proj_wmma<<<4096, 128, 0, stream>>>(pooled, pw, pb, out);

  // ---- sort 2: order = argsort(op_logits) ----
  k_sort_init<<<N2 / 256, 256, 0, stream>>>(out, k2A, v2A, N2);
  ki = k2A; vi = v2A; ko = k2B; vo = v2B;
  for (int pass = 0; pass < 4; pass++) {
    int shift = pass * 8;
    k_sort_hist<<<T2, 256, 0, stream>>>(ki, hist, shift, T2);
    k_scan<<<1, 1024, 0, stream>>>(hist, 256 * T2);
    k_sort_scatter<<<T2, 256, 0, stream>>>(ki, vi, ko, vo, hist, shift, T2);
    unsigned* tk = ki; ki = ko; ko = tk;
    unsigned* tv = vi; vi = vo; vo = tv;
  }
  unsigned* order = vi;

  // ---- penalties -> d_out[N2], d_out[N2+1] ----
  k_penalty<<<1024, 256, 0, stream>>>(order, addr, psum);
  k_pen_fin<<<1, 256, 0, stream>>>(psum, out + N2);

  (void)in_sizes; (void)n_in; (void)out_size; (void)ws_size;
}
